// Net_23587960389976
// MI455X (gfx1250) — matrix operand order repaired and checked by reading, not compiled, not run on hardware
//
#include <hip/hip_runtime.h>
#include <hip/hip_bf16.h>

// ---------------------------------------------------------------------------
// SplineConv GNN (dim=1, K=2, degree=1, mean aggr), 2 layers + ELU + log_softmax
// Layer-1 transform fused as x[N,1433] @ Wcat[1433,48] via v_wmma_f32_16x16x32_bf16
// ---------------------------------------------------------------------------

#define F_IN   1433
#define K_PAD  1440         // F_IN padded up to a multiple of 32 (zero-filled)
#define F_HID  16
#define F_OUT  7
#define NCOL1  48           // [ W1[0] | W1[1] | root1 ] columns
#define NCOL2  21           // [ W2[0] | W2[1] | root2 ] columns

typedef __attribute__((ext_vector_type(16))) __bf16         v16bf;
typedef __attribute__((ext_vector_type(8)))  float          v8f;
typedef __attribute__((ext_vector_type(8)))  unsigned short ushort8;

union BFrag {                 // 16 bf16 lane-fragment
    __bf16         b[16];
    unsigned short u[16];
    ushort8        h[2];
    v16bf          v;
};

__device__ __forceinline__ unsigned short f2bf(float f) {
    unsigned int x = __float_as_uint(f);
    unsigned int r = x + 0x7fffu + ((x >> 16) & 1u);   // round-to-nearest-even
    return (unsigned short)(r >> 16);
}

// --- weight prep: Wcat^T [48][1440] in bf16, zero-padded past K=1433 ---------
__global__ void spline_prep_w(const float* __restrict__ W1,
                              const float* __restrict__ root1,
                              unsigned short* __restrict__ WbT) {
    int t = blockIdx.x * blockDim.x + threadIdx.x;
    if (t >= NCOL1 * K_PAD) return;
    int n = t / K_PAD, kk = t % K_PAD;
    float v = 0.f;
    if (kk < F_IN) {
        if (n < 16)      v = W1[(0 * F_IN + kk) * F_HID + n];
        else if (n < 32) v = W1[(1 * F_IN + kk) * F_HID + (n - 16)];
        else             v = root1[kk * F_HID + (n - 32)];
    }
    WbT[t] = f2bf(v);
}

// --- layer-1 fused GEMM: H1[N,48] = x @ Wcat ------------------------------
// 1 wave = one 16-row tile x all 48 cols (3 accumulators). x read exactly once.
__global__ __launch_bounds__(128)
void spline_gemm1(const float* __restrict__ x,
                  const unsigned short* __restrict__ WbT,
                  float* __restrict__ H1, int nTiles, int nNodes) {
    const int lane = threadIdx.x & 31;
    const int wave = threadIdx.x >> 5;
    const int tile = blockIdx.x * 4 + wave;
    if (tile >= nTiles) return;                 // wave-uniform

    const int rowBase = tile * 16;
    const int hf = lane >> 4;                   // K-half selector
    const int lp = lane & 15;                   // row (A) / col (B,D) in tile
    int row = rowBase + lp;
    if (row >= nNodes) row = nNodes - 1;        // clamp; store is guarded
    const float* rp = x + (size_t)row * F_IN;

    // B fragment base pointers (col-tile 0/1/2); layout [48][1440] bf16,
    // row stride 2880 B and hf*16 elem = 32 B offset -> 16B-aligned ushort8.
    const ushort8* b0p = (const ushort8*)(WbT + (size_t)(lp)      * K_PAD + hf * 16);
    const ushort8* b1p = (const ushort8*)(WbT + (size_t)(16 + lp) * K_PAD + hf * 16);
    const ushort8* b2p = (const ushort8*)(WbT + (size_t)(32 + lp) * K_PAD + hf * 16);

    v8f c0 = {}, c1 = {}, c2 = {};

    int k0 = 0;
    #pragma unroll 2
    for (; k0 + 32 <= F_IN; k0 += 32) {         // 44 full K-steps
        __builtin_prefetch(rp + k0 + 64, 0, 0); // speculative look-ahead
        BFrag a;
        const int kA = k0 + hf * 16;
        #pragma unroll
        for (int i = 0; i < 8; ++i) a.b[i]     = (__bf16)rp[kA + i];       // v_cvt_pk_bf16_f32
        #pragma unroll
        for (int i = 0; i < 8; ++i) a.b[8 + i] = (__bf16)rp[kA + 8 + i];

        const int k8 = k0 >> 3;                 // ushort8 units
        BFrag b0, b1, b2;
        b0.h[0] = b0p[k8]; b0.h[1] = b0p[k8 + 1];
        b1.h[0] = b1p[k8]; b1.h[1] = b1p[k8 + 1];
        b2.h[0] = b2p[k8]; b2.h[1] = b2p[k8 + 1];

        c0 = __builtin_amdgcn_wmma_f32_16x16x32_bf16(false, a.v, false, b0.v, (short)0, c0, false, false);
        c1 = __builtin_amdgcn_wmma_f32_16x16x32_bf16(false, a.v, false, b1.v, (short)0, c1, false, false);
        c2 = __builtin_amdgcn_wmma_f32_16x16x32_bf16(false, a.v, false, b2.v, (short)0, c2, false, false);
    }
    {   // tail K-step (k0 = 1408), A guarded, B zero-padded to K_PAD
        BFrag a;
        const int kA = k0 + hf * 16;
        #pragma unroll
        for (int i = 0; i < 8; ++i) a.b[i]     = (kA + i      < F_IN) ? (__bf16)rp[kA + i]      : (__bf16)0.f;
        #pragma unroll
        for (int i = 0; i < 8; ++i) a.b[8 + i] = (kA + 8 + i  < F_IN) ? (__bf16)rp[kA + 8 + i]  : (__bf16)0.f;
        const int k8 = k0 >> 3;
        BFrag b0, b1, b2;
        b0.h[0] = b0p[k8]; b0.h[1] = b0p[k8 + 1];
        b1.h[0] = b1p[k8]; b1.h[1] = b1p[k8 + 1];
        b2.h[0] = b2p[k8]; b2.h[1] = b2p[k8 + 1];
        c0 = __builtin_amdgcn_wmma_f32_16x16x32_bf16(false, a.v, false, b0.v, (short)0, c0, false, false);
        c1 = __builtin_amdgcn_wmma_f32_16x16x32_bf16(false, a.v, false, b1.v, (short)0, c1, false, false);
        c2 = __builtin_amdgcn_wmma_f32_16x16x32_bf16(false, a.v, false, b2.v, (short)0, c2, false, false);
    }

    // D layout: VGPR r, lane -> row = rowBase + r + 8*hf, col = colTile*16 + lp
    if (rowBase + 16 <= nNodes) {               // full tile (always for N % 16 == 0)
        float* op = H1 + (size_t)(rowBase + hf * 8) * NCOL1 + lp;
        #pragma unroll
        for (int r = 0; r < 8; ++r) {
            op[(size_t)r * NCOL1 + 0]  = c0[r];
            op[(size_t)r * NCOL1 + 16] = c1[r];
            op[(size_t)r * NCOL1 + 32] = c2[r];
        }
    } else {
        #pragma unroll
        for (int r = 0; r < 8; ++r) {
            const int orow = rowBase + hf * 8 + r;
            if (orow < nNodes) {
                float* op = H1 + (size_t)orow * NCOL1 + lp;
                op[0]  = c0[r];
                op[16] = c1[r];
                op[32] = c2[r];
            }
        }
    }
}

// --- layer-1 edge message + scatter: agg1[dst] += (1-u)*h0[src] + u*h1[src] --
__global__ void spline_edge1(const int* __restrict__ ei,
                             const float* __restrict__ ea,
                             const float* __restrict__ H1,
                             float* __restrict__ agg1,
                             float* __restrict__ deg, int nEdges) {
    int e = blockIdx.x * blockDim.x + threadIdx.x;
    if (e >= nEdges) return;
    const int src = ei[e], dst = ei[nEdges + e];
    const float t = ea[e], s = 1.f - t;
    const float4* h0 = (const float4*)(H1 + (size_t)src * NCOL1);       // cols 0..15
    const float4* h1 = (const float4*)(H1 + (size_t)src * NCOL1 + 16);  // cols 16..31
    float* ap = agg1 + (size_t)dst * F_HID;
    #pragma unroll
    for (int q = 0; q < 4; ++q) {
        float4 a = h0[q], b = h1[q];
        atomicAdd(&ap[4 * q + 0], s * a.x + t * b.x);
        atomicAdd(&ap[4 * q + 1], s * a.y + t * b.y);
        atomicAdd(&ap[4 * q + 2], s * a.z + t * b.z);
        atomicAdd(&ap[4 * q + 3], s * a.w + t * b.w);
    }
    atomicAdd(&deg[dst], 1.f);
}

// --- layer-1 finalize: mean + root (H1 cols 32..47) + bias + ELU ------------
__global__ void spline_fin1(const float* __restrict__ agg1,
                            const float* __restrict__ deg,
                            const float* __restrict__ H1,
                            const float* __restrict__ b1,
                            float* __restrict__ hact, int nNodes) {
    int n = blockIdx.x * blockDim.x + threadIdx.x;
    if (n >= nNodes) return;
    float d = deg[n]; d = d < 1.f ? 1.f : d;
    const float inv = 1.f / d;
    #pragma unroll
    for (int j = 0; j < F_HID; ++j) {
        float v = agg1[(size_t)n * F_HID + j] * inv
                + H1[(size_t)n * NCOL1 + 32 + j] + b1[j];
        hact[(size_t)n * F_HID + j] = v > 0.f ? v : (expf(v) - 1.f);   // ELU
    }
}

// --- layer-2 transform: H2[N,21] = hact @ [W2[0]|W2[1]|root2] (tiny K=16) ---
__global__ void spline_lin2(const float* __restrict__ hact,
                            const float* __restrict__ W2,
                            const float* __restrict__ root2,
                            float* __restrict__ H2, int nNodes) {
    __shared__ float w[F_HID * NCOL2];
    for (int t = threadIdx.x; t < F_HID * NCOL2; t += blockDim.x) {
        int i = t / NCOL2, o = t % NCOL2;
        float v;
        if (o < 7)       v = W2[(0 * F_HID + i) * F_OUT + o];
        else if (o < 14) v = W2[(1 * F_HID + i) * F_OUT + (o - 7)];
        else             v = root2[i * F_OUT + (o - 14)];
        w[t] = v;
    }
    __syncthreads();
    int n = blockIdx.x * blockDim.x + threadIdx.x;
    if (n >= nNodes) return;
    float hv[F_HID];
    const float* hp = hact + (size_t)n * F_HID;
    #pragma unroll
    for (int i = 0; i < F_HID; ++i) hv[i] = hp[i];
    float acc[NCOL2];
    #pragma unroll
    for (int o = 0; o < NCOL2; ++o) acc[o] = 0.f;
    #pragma unroll
    for (int i = 0; i < F_HID; ++i)
        #pragma unroll
        for (int o = 0; o < NCOL2; ++o)
            acc[o] = fmaf(hv[i], w[i * NCOL2 + o], acc[o]);
    float* op = H2 + (size_t)n * NCOL2;
    #pragma unroll
    for (int o = 0; o < NCOL2; ++o) op[o] = acc[o];
}

// --- layer-2 edge message + scatter ----------------------------------------
__global__ void spline_edge2(const int* __restrict__ ei,
                             const float* __restrict__ ea,
                             const float* __restrict__ H2,
                             float* __restrict__ agg2, int nEdges) {
    int e = blockIdx.x * blockDim.x + threadIdx.x;
    if (e >= nEdges) return;
    const int src = ei[e], dst = ei[nEdges + e];
    const float t = ea[e], s = 1.f - t;
    const float* hp = H2 + (size_t)src * NCOL2;
    float* ap = agg2 + (size_t)dst * F_OUT;
    #pragma unroll
    for (int j = 0; j < F_OUT; ++j)
        atomicAdd(&ap[j], s * hp[j] + t * hp[7 + j]);
}

// --- layer-2 finalize: mean + root + bias + log_softmax ---------------------
__global__ void spline_fin2(const float* __restrict__ agg2,
                            const float* __restrict__ deg,
                            const float* __restrict__ H2,
                            const float* __restrict__ b2,
                            float* __restrict__ out, int nNodes) {
    int n = blockIdx.x * blockDim.x + threadIdx.x;
    if (n >= nNodes) return;
    float d = deg[n]; d = d < 1.f ? 1.f : d;
    const float inv = 1.f / d;
    float v[F_OUT];
    #pragma unroll
    for (int j = 0; j < F_OUT; ++j)
        v[j] = agg2[(size_t)n * F_OUT + j] * inv
             + H2[(size_t)n * NCOL2 + 14 + j] + b2[j];
    float m = v[0];
    #pragma unroll
    for (int j = 1; j < F_OUT; ++j) m = fmaxf(m, v[j]);
    float s = 0.f;
    #pragma unroll
    for (int j = 0; j < F_OUT; ++j) s += expf(v[j] - m);
    const float ls = logf(s);
    #pragma unroll
    for (int j = 0; j < F_OUT; ++j)
        out[(size_t)n * F_OUT + j] = v[j] - m - ls;
}

// ---------------------------------------------------------------------------
extern "C" void kernel_launch(void* const* d_in, const int* in_sizes, int n_in,
                              void* d_out, int out_size, void* d_ws, size_t ws_size,
                              hipStream_t stream) {
    const float* x     = (const float*)d_in[0];
    const float* ea    = (const float*)d_in[1];
    const int*   ei    = (const int*)  d_in[2];
    const float* W1    = (const float*)d_in[3];
    const float* root1 = (const float*)d_in[4];
    const float* b1    = (const float*)d_in[5];
    const float* W2    = (const float*)d_in[6];
    const float* root2 = (const float*)d_in[7];
    const float* b2    = (const float*)d_in[8];
    float* out = (float*)d_out;

    const int nNodes = in_sizes[0] / F_IN;
    const int nEdges = in_sizes[1];

    // workspace carve (256B-aligned slabs)
    char* p = (char*)d_ws;
    auto carve = [&](size_t bytes) {
        char* r = p; p += (bytes + 255) & ~(size_t)255; return r;
    };
    unsigned short* WbT = (unsigned short*)carve((size_t)NCOL1 * K_PAD * 2);
    float* H1   = (float*)carve((size_t)nNodes * NCOL1 * 4);
    float* agg1 = (float*)carve((size_t)nNodes * F_HID * 4);
    float* deg  = (float*)carve((size_t)nNodes * 4);
    float* hact = (float*)carve((size_t)nNodes * F_HID * 4);
    float* H2   = (float*)carve((size_t)nNodes * NCOL2 * 4);
    float* agg2 = (float*)carve((size_t)nNodes * F_OUT * 4);

    // deterministic per-call zeroing of accumulators (capture-safe)
    hipMemsetAsync(agg1, 0, (size_t)nNodes * F_HID * 4, stream);
    hipMemsetAsync(deg,  0, (size_t)nNodes * 4,         stream);
    hipMemsetAsync(agg2, 0, (size_t)nNodes * F_OUT * 4, stream);

    spline_prep_w<<<(NCOL1 * K_PAD + 255) / 256, 256, 0, stream>>>(W1, root1, WbT);

    const int nTiles = (nNodes + 15) / 16;
    spline_gemm1<<<(nTiles + 3) / 4, 128, 0, stream>>>(x, WbT, H1, nTiles, nNodes);

    spline_edge1<<<(nEdges + 255) / 256, 256, 0, stream>>>(ei, ea, H1, agg1, deg, nEdges);
    spline_fin1 <<<(nNodes + 255) / 256, 256, 0, stream>>>(agg1, deg, H1, b1, hact, nNodes);
    spline_lin2 <<<(nNodes + 255) / 256, 256, 0, stream>>>(hact, W2, root2, H2, nNodes);
    spline_edge2<<<(nEdges + 255) / 256, 256, 0, stream>>>(ei, ea, H2, agg2, nEdges);
    spline_fin2 <<<(nNodes + 255) / 256, 256, 0, stream>>>(agg2, deg, H2, b2, out, nNodes);
}
